// local_purificaiton_module_16527034155655
// MI455X (gfx1250) — matrix operand order repaired and checked by reading, not compiled, not run on hardware
//
// MI455X (gfx1250) implementation of the lead-wise transformer + purification top-k.
// All matmuls via v_wmma_f32_16x16x32_bf16 (bf16 operands, f32 accum).
// Pipeline (per call): add_pos -> [QKV gemms -> attention -> O-proj -> LN+res ->
//   4x(MLP1 slice gemm (gelu) -> MLP2 slice gemm (accum)) -> LN+res] x2 ->
//   final LN -> pf0 gemm(gelu) -> pf1 gemm(gelu) -> score+bitonic-topk+gather.
// Workspace requirement: 5 buffers of 480*256*128 f32 = ~315 MB.
// Round-3 fix: accumulator array was runtime-indexed (cb < nb with runtime nb),
// so LLVM demoted it to scratch. Column-tile count is now a template constant
// (NB=8 / NB=2), cb loop fully unrolls, accumulators stay in VGPRs.

#include <hip/hip_runtime.h>

#define SEQS 480   // 32 batch * 15 leads
#define TOK  256   // tokens per lead
#define EMB  128

static constexpr size_t SBUF = (size_t)SEQS * TOK * EMB; // 15,728,640 floats per buffer

typedef __attribute__((ext_vector_type(16))) __bf16        bf16x16;
typedef __attribute__((ext_vector_type(8)))  float         f32x8;
typedef __attribute__((ext_vector_type(4)))  unsigned int  u32x4;

// ---------- helpers ----------

__device__ __forceinline__ unsigned short f2bf(float f) {
  union { __bf16 h; unsigned short u; } cv;
  cv.h = (__bf16)f;                    // hardware cvt, round-to-nearest-even
  return cv.u;
}

__device__ __forceinline__ f32x8 zero8() {
  f32x8 z;
#pragma unroll
  for (int i = 0; i < 8; ++i) z[i] = 0.0f;
  return z;
}

union FragU { bf16x16 v; u32x4 q[2]; unsigned short u[16]; };

// A-operand fragment (16x32 bf16): lanes 0-15 rows m, K 0..15; lanes 16-31 K 16..31
__device__ __forceinline__ bf16x16 lds_fragA(const unsigned short* s, int row, int ld, int k0) {
  const int l = threadIdx.x & 31;
  const unsigned short* p = s + (size_t)(row + (l & 15)) * ld + k0 + (l >> 4) * 16;
  FragU f;
  f.q[0] = *(const u32x4*)(p);
  f.q[1] = *(const u32x4*)(p + 8);
  return f.v;
}

// B-operand fragment (32x16 bf16, lane n = column): lanes 0-15 K 0..15, lanes 16-31 K 16..31
__device__ __forceinline__ bf16x16 lds_fragB(const unsigned short* s, int row, int ld, int k0) {
  const int l = threadIdx.x & 31;
  const unsigned short* p = s + (size_t)(row + (l & 15)) * ld + k0 + (l >> 4) * 16;
  FragU f;
  f.q[0] = *(const u32x4*)(p);
  f.q[1] = *(const u32x4*)(p + 8);
  return f.v;
}

__device__ __forceinline__ bf16x16 gmem_fragA(const float* g, int row, int ld, int k0) {
  const int l = threadIdx.x & 31;
  const float* p = g + (size_t)(row + (l & 15)) * ld + k0 + (l >> 4) * 16;
  FragU f;
#pragma unroll
  for (int i = 0; i < 8; ++i) f.u[i]     = f2bf(p[i]);
#pragma unroll
  for (int i = 0; i < 8; ++i) f.u[8 + i] = f2bf(p[8 + i]);
  return f.v;
}

__device__ __forceinline__ bf16x16 gmem_fragB(const float* g, int row, int ld, int k0) {
  const int l = threadIdx.x & 31;
  const float* p = g + (size_t)(row + (l & 15)) * ld + k0 + (l >> 4) * 16;
  FragU f;
#pragma unroll
  for (int i = 0; i < 16; ++i) f.u[i] = f2bf(p[i]);
  return f.v;
}

__device__ __forceinline__ f32x8 wmma_bf16(bf16x16 a, bf16x16 b, f32x8 c) {
  return __builtin_amdgcn_wmma_f32_16x16x32_bf16(false, a, false, b, (short)0, c, false, false);
}

__device__ __forceinline__ float gelu_exact(float x) {
  return 0.5f * x * (1.0f + erff(x * 0.70710678118654752f));
}

__device__ __forceinline__ float wave_sum32(float s) {
#pragma unroll
  for (int m = 1; m < 32; m <<= 1) s += __shfl_xor(s, m, 32);
  return s;
}
__device__ __forceinline__ float half_sum16(float s) {
#pragma unroll
  for (int m = 1; m < 16; m <<= 1) s += __shfl_xor(s, m, 16);
  return s;
}
__device__ __forceinline__ float half_max16(float s) {
#pragma unroll
  for (int m = 1; m < 16; m <<= 1) s = fmaxf(s, __shfl_xor(s, m, 16));
  return s;
}

// ---------- kernels ----------

__global__ __launch_bounds__(256) void add_pos_kernel(const float4* __restrict__ a,
                                                      const float4* __restrict__ b,
                                                      float4* __restrict__ o, int n4) {
  int i = blockIdx.x * 256 + threadIdx.x;
  if (i < n4) {
    float4 x = a[i], y = b[i];
    o[i] = make_float4(x.x + y.x, x.y + y.y, x.z + y.z, x.w + y.w);
  }
}

// Batched C[b] = A[b] @ W^T + bias (torch Linear), N = NB*16 <= 128 (compile-time).
// Grid (SEQS, 2): each block does a 128-row half; each wave 16 rows x N cols.
// NB is a template constant so the accumulator array fully scalarizes to VGPRs.
template <int NB>
__global__ __launch_bounds__(256, 1) void gemm_bias_kernel(
    const float* __restrict__ A, int ald,
    const float* __restrict__ W, int wld,       // pre-offset to the N-slice
    const float* __restrict__ bias,             // pre-offset, may be null
    float* __restrict__ C, int cld,
    int K, int gelu, int accum) {
  __shared__ unsigned short Ws[128 * 136];
  const int b = blockIdx.x;
  const int mh = blockIdx.y;                    // 128-row half of the sequence
  const int wave = threadIdx.x >> 5;
  const float* Ab = A + ((size_t)b * TOK + mh * 128) * ald;
  float* Cb = C + ((size_t)b * TOK + mh * 128) * cld;
  const int l = threadIdx.x & 31, n = l & 15, ro = (l >> 4) * 8;
  const int r0 = wave * 16;
  constexpr int Nc = NB * 16;

  f32x8 acc[NB];
#pragma unroll
  for (int cb = 0; cb < NB; ++cb) acc[cb] = zero8();

  for (int kc = 0; kc < K; kc += 128) {
    __syncthreads();
    // stage W chunk (Nc x 128, f32 -> bf16) into LDS, ld=136 (conflict-free b128 reads)
    for (int i = threadIdx.x; i < Nc * 32; i += 256) {
      int idx = i * 4; int r = idx >> 7; int c = idx & 127;
      float4 f = *(const float4*)(W + (size_t)r * wld + kc + c);
      unsigned short* d = Ws + r * 136 + c;
      d[0] = f2bf(f.x); d[1] = f2bf(f.y); d[2] = f2bf(f.z); d[3] = f2bf(f.w);
    }
    __syncthreads();
#pragma unroll
    for (int kb = 0; kb < 4; ++kb) {
      bf16x16 a = gmem_fragA(Ab, r0, ald, kc + kb * 32);
#pragma unroll
      for (int cb = 0; cb < NB; ++cb) {
        bf16x16 w = lds_fragB(Ws, cb * 16, 136, kb * 32);
        acc[cb] = wmma_bf16(a, w, acc[cb]);
      }
    }
  }
  // epilogue (each store instruction covers 2 rows x 16 consecutive floats)
#pragma unroll
  for (int cb = 0; cb < NB; ++cb) {
    int col = cb * 16 + n;
    float bv = bias ? bias[col] : 0.0f;
#pragma unroll
    for (int r = 0; r < 8; ++r) {
      float x = acc[cb][r] + bv;
      size_t o = (size_t)(r0 + ro + r) * cld + col;
      if (accum) x += Cb[o];
      if (gelu) x = gelu_exact(x);
      Cb[o] = x;
    }
  }
}

// Full attention for one (sequence, head), two-pass streaming softmax:
// pass 1 computes QK^T tile-by-tile for row maxes (one f32 tile live at a time),
// pass 2 recomputes each tile, exponentiates, accumulates row sums, stages the
// un-normalized P through per-wave LDS, and WMMAs into the O accumulator;
// O is scaled by 1/rowsum at the end.
__global__ __launch_bounds__(256, 1) void attn_kernel(const float* __restrict__ Q,
                                                      const float* __restrict__ K,
                                                      const float* __restrict__ V,
                                                      float* __restrict__ O) {
  __shared__ unsigned short vts[64 * 264];        // V^T (dh x 256, padded)
  __shared__ unsigned short ps[8 * 16 * 40];      // per-wave 16x32 P stage
  const int b = blockIdx.x, h = blockIdx.y;
  const float* Qb = Q + ((size_t)b * TOK) * EMB + h * 64;
  const float* Kb = K + ((size_t)b * TOK) * EMB + h * 64;
  const float* Vb = V + ((size_t)b * TOK) * EMB + h * 64;
  float* Ob = O + ((size_t)b * TOK) * EMB + h * 64;

  for (int i = threadIdx.x; i < TOK * 16; i += 256) {  // transpose-stage V
    int j = i >> 4; int d4 = (i & 15) * 4;
    float4 f = *(const float4*)(Vb + (size_t)j * EMB + d4);
    vts[(d4 + 0) * 264 + j] = f2bf(f.x);
    vts[(d4 + 1) * 264 + j] = f2bf(f.y);
    vts[(d4 + 2) * 264 + j] = f2bf(f.z);
    vts[(d4 + 3) * 264 + j] = f2bf(f.w);
  }
  __syncthreads();

  const int wave = threadIdx.x >> 5;
  const int l = threadIdx.x & 31, n = l & 15, ro = (l >> 4) * 8;
  unsigned short* myps = ps + wave * 16 * 40;
  const int r0 = wave * 32;

  for (int rb = 0; rb < 2; ++rb) {
    const int row0 = r0 + rb * 16;
    bf16x16 qf0 = gmem_fragA(Qb, row0, EMB, 0);
    bf16x16 qf1 = gmem_fragA(Qb, row0, EMB, 32);

    // pass 1: per-row max of scores
    float mrow[8];
#pragma unroll
    for (int r = 0; r < 8; ++r) mrow[r] = -3.0e38f;
    for (int jt = 0; jt < 16; ++jt) {
      f32x8 dt = zero8();
      dt = wmma_bf16(qf0, gmem_fragB(Kb, jt * 16, EMB, 0), dt);
      dt = wmma_bf16(qf1, gmem_fragB(Kb, jt * 16, EMB, 32), dt);
#pragma unroll
      for (int r = 0; r < 8; ++r) mrow[r] = fmaxf(mrow[r], dt[r]);
    }
#pragma unroll
    for (int r = 0; r < 8; ++r) mrow[r] = half_max16(mrow[r]);

    // pass 2: exp, row-sum, and un-normalized P @ V
    float rs[8];
#pragma unroll
    for (int r = 0; r < 8; ++r) rs[r] = 0.0f;
    f32x8 oacc[4];
#pragma unroll
    for (int c = 0; c < 4; ++c) oacc[c] = zero8();

    for (int jt = 0; jt < 16; ++jt) {
      f32x8 dt = zero8();
      dt = wmma_bf16(qf0, gmem_fragB(Kb, jt * 16, EMB, 0), dt);
      dt = wmma_bf16(qf1, gmem_fragB(Kb, jt * 16, EMB, 32), dt);
      const int s2 = jt & 1;
#pragma unroll
      for (int r = 0; r < 8; ++r) {
        float p = __expf((dt[r] - mrow[r]) * 0.125f);   // scale = dh^-0.5
        rs[r] += p;
        myps[(ro + r) * 40 + s2 * 16 + n] = f2bf(p);
      }
      if (s2) {   // two 16x16 tiles staged -> one 16x32 A-fragment chunk
        asm volatile("s_wait_dscnt 0" ::: "memory");
        bf16x16 pf = lds_fragA(myps, 0, 40, 0);
#pragma unroll
        for (int ct = 0; ct < 4; ++ct) {
          bf16x16 vf = lds_fragB(vts, ct * 16, 264, (jt >> 1) * 32);
          oacc[ct] = wmma_bf16(pf, vf, oacc[ct]);
        }
      }
    }
#pragma unroll
    for (int r = 0; r < 8; ++r) rs[r] = 1.0f / half_sum16(rs[r]);
#pragma unroll
    for (int ct = 0; ct < 4; ++ct) {
      int col = ct * 16 + n;
#pragma unroll
      for (int r = 0; r < 8; ++r)
        Ob[(size_t)(row0 + ro + r) * EMB + col] = oacc[ct][r] * rs[r];
    }
  }
}

// out = (res ? res : 0) + LayerNorm(y) * gamma + beta ; one wave per 128-wide row.
__global__ __launch_bounds__(256) void ln_kernel(const float* __restrict__ y,
                                                 const float* __restrict__ gam,
                                                 const float* __restrict__ bet,
                                                 const float* __restrict__ res,
                                                 float* __restrict__ out) {
  const int row = blockIdx.x * 8 + (threadIdx.x >> 5);
  const int l = threadIdx.x & 31;
  const float* yr = y + (size_t)row * EMB;
  float4 v = *(const float4*)(yr + l * 4);
  float mean = wave_sum32(v.x + v.y + v.z + v.w) * (1.0f / 128.0f);
  float dx = v.x - mean, dy = v.y - mean, dz = v.z - mean, dw = v.w - mean;
  float var = wave_sum32(dx * dx + dy * dy + dz * dz + dw * dw) * (1.0f / 128.0f);
  float inv = rsqrtf(var + 1e-5f);
  int c = l * 4;
  float4 o;
  o.x = dx * inv * gam[c + 0] + bet[c + 0];
  o.y = dy * inv * gam[c + 1] + bet[c + 1];
  o.z = dz * inv * gam[c + 2] + bet[c + 2];
  o.w = dw * inv * gam[c + 3] + bet[c + 3];
  if (res) {
    float4 r = *(const float4*)(res + (size_t)row * EMB + c);
    o.x += r.x; o.y += r.y; o.z += r.z; o.w += r.w;
  }
  *(float4*)(out + (size_t)row * EMB + c) = o;
}

// pf3 dot -> bitonic sort 256 by (score desc, idx asc) -> gather 128 rows of x and pos.
// (softmax is monotonic; only indices are consumed, so it is skipped)
__global__ __launch_bounds__(256) void topk_gather_kernel(
    const float* __restrict__ S32, const float* __restrict__ w3, const float* __restrict__ b3,
    const float* __restrict__ XN, const float* __restrict__ POS,
    float* __restrict__ XOUT, float* __restrict__ POUT) {
  __shared__ float key[256];
  __shared__ int   kid[256];
  const int seq = blockIdx.x, t = threadIdx.x;
  const float* sr = S32 + ((size_t)seq * TOK + t) * 32;
  float acc = b3[0];
#pragma unroll
  for (int i = 0; i < 32; ++i) acc += sr[i] * w3[i];
  key[t] = acc; kid[t] = t;
  __syncthreads();
  for (int k = 2; k <= 256; k <<= 1)
    for (int j = k >> 1; j > 0; j >>= 1) {
      int ixj = t ^ j;
      if (ixj > t) {
        float ka = key[t], kb = key[ixj];
        int ia = kid[t], ib = kid[ixj];
        bool before = (ka > kb) || (ka == kb && ia < ib);
        bool want = ((t & k) == 0) ? before : !before;
        if (!want) { key[t] = kb; key[ixj] = ka; kid[t] = ib; kid[ixj] = ia; }
      }
      __syncthreads();
    }
  const int bsi = seq / 15, lead = seq % 15;
  for (int i = t; i < 128 * 32; i += 256) {
    int r = i >> 5; int c4 = (i & 31) * 4;
    int tok = kid[r];
    size_t orow = (size_t)bsi * 1920 + lead * 128 + r;
    *(float4*)(XOUT + orow * EMB + c4) =
        *(const float4*)(XN + ((size_t)seq * TOK + tok) * EMB + c4);
    *(float4*)(POUT + orow * EMB + c4) =
        *(const float4*)(POS + ((size_t)bsi * 3840 + lead * TOK + tok) * EMB + c4);
  }
}

// ---------- host orchestration ----------

extern "C" void kernel_launch(void* const* d_in, const int* in_sizes, int n_in,
                              void* d_out, int out_size, void* d_ws, size_t ws_size,
                              hipStream_t stream) {
  (void)in_sizes; (void)out_size;
  if (n_in < 42) return;
  if (ws_size < 5ull * SBUF * sizeof(float)) return;

  const float* x   = (const float*)d_in[0];
  const float* pos = (const float*)d_in[1];
  float* ws  = (float*)d_ws;
  float* xx  = ws + 0 * SBUF;  // residual stream
  float* bA  = ws + 1 * SBUF;  // q / y / xn
  float* bB  = ws + 2 * SBUF;  // k / h_chunk / s128
  float* bC  = ws + 3 * SBUF;  // v / y_acc / s32
  float* bD  = ws + 4 * SBUF;  // attention O

  const dim3 gg(SEQS, 2);

  add_pos_kernel<<<15360, 256, 0, stream>>>((const float4*)x, (const float4*)pos,
                                            (float4*)xx, (int)(SBUF / 4));

  for (int layer = 0; layer < 2; ++layer) {
    const int p = 2 + layer * 16;
    const float *wq = (const float*)d_in[p + 0],  *bq = (const float*)d_in[p + 1];
    const float *wk = (const float*)d_in[p + 2],  *bk = (const float*)d_in[p + 3];
    const float *wv = (const float*)d_in[p + 4],  *bv = (const float*)d_in[p + 5];
    const float *wo = (const float*)d_in[p + 6],  *bo = (const float*)d_in[p + 7];
    const float *n1s = (const float*)d_in[p + 8], *n1b = (const float*)d_in[p + 9];
    const float *n2s = (const float*)d_in[p + 10], *n2b = (const float*)d_in[p + 11];
    const float *w1 = (const float*)d_in[p + 12], *b1 = (const float*)d_in[p + 13];
    const float *w2 = (const float*)d_in[p + 14], *b2 = (const float*)d_in[p + 15];

    gemm_bias_kernel<8><<<gg, 256, 0, stream>>>(xx, 128, wq, 128, bq, bA, 128, 128, 0, 0);
    gemm_bias_kernel<8><<<gg, 256, 0, stream>>>(xx, 128, wk, 128, bk, bB, 128, 128, 0, 0);
    gemm_bias_kernel<8><<<gg, 256, 0, stream>>>(xx, 128, wv, 128, bv, bC, 128, 128, 0, 0);
    attn_kernel<<<dim3(SEQS, 2), 256, 0, stream>>>(bA, bB, bC, bD);
    gemm_bias_kernel<8><<<gg, 256, 0, stream>>>(bD, 128, wo, 128, bo, bA, 128, 128, 0, 0);
    ln_kernel<<<15360, 256, 0, stream>>>(bA, n1s, n1b, xx, xx);   // x += LN(attn)

    for (int cc = 0; cc < 4; ++cc) {  // MLP hidden=512 sliced into 4x128
      gemm_bias_kernel<8><<<gg, 256, 0, stream>>>(xx, 128, w1 + (size_t)cc * 128 * 128, 128,
                                                  b1 + cc * 128, bB, 128, 128, 1, 0);
      gemm_bias_kernel<8><<<gg, 256, 0, stream>>>(bB, 128, w2 + cc * 128, 512,
                                                  cc == 0 ? b2 : nullptr, bC, 128, 128,
                                                  0, cc > 0 ? 1 : 0);
    }
    ln_kernel<<<15360, 256, 0, stream>>>(bC, n2s, n2b, xx, xx);   // x += LN(mlp)
  }

  const float *ens = (const float*)d_in[34], *enb = (const float*)d_in[35];
  const float *pf0w = (const float*)d_in[36], *pf0b = (const float*)d_in[37];
  const float *pf1w = (const float*)d_in[38], *pf1b = (const float*)d_in[39];
  const float *pf3w = (const float*)d_in[40], *pf3b = (const float*)d_in[41];

  ln_kernel<<<15360, 256, 0, stream>>>(xx, ens, enb, nullptr, bA);              // xn
  gemm_bias_kernel<8><<<gg, 256, 0, stream>>>(bA, 128, pf0w, 128, pf0b, bB, 128, 128, 1, 0);
  gemm_bias_kernel<2><<<gg, 256, 0, stream>>>(bB, 128, pf1w, 128, pf1b, bC, 32, 128, 1, 0);

  float* xout = (float*)d_out;
  float* pout = xout + (size_t)32 * 1920 * 128;
  topk_gather_kernel<<<SEQS, 256, 0, stream>>>(bC, pf3w, pf3b, bA, pos, xout, pout);
}
